// ConvSwinTransformerBlock_23759759081713
// MI455X (gfx1250) — matrix operand order repaired and checked by reading, not compiled, not run on hardware
//
#include <hip/hip_runtime.h>
#include <hip/hip_bf16.h>
#include <math.h>

typedef _Float16 v16h __attribute__((ext_vector_type(16)));
typedef _Float16 v8h  __attribute__((ext_vector_type(8)));
typedef float    v8f  __attribute__((ext_vector_type(8)));

__device__ __forceinline__ float gelu_exact(float x) {
    return 0.5f * x * (1.f + erff(x * 0.70710678118654752f));
}
__device__ __forceinline__ float wred_sum32(float v) {
    #pragma unroll
    for (int o = 16; o; o >>= 1) v += __shfl_xor(v, o, 32);
    return v;
}
__device__ __forceinline__ v8f wmma_f16(v16h a, v16h b, v8f c) {
    return __builtin_amdgcn_wmma_f32_16x16x32_f16(false, a, false, b, (short)0, c, false, false);
}
// A-fragment (16x32 f16): per lane, halves 0..15 = K 16*hi..16*hi+15
__device__ __forceinline__ v16h load_afrag(const _Float16* row, int hi) {
    v8h l0 = *(const v8h*)(row + 16 * hi);
    v8h l1 = *(const v8h*)(row + 16 * hi + 8);
    v16h a;
    #pragma unroll
    for (int t = 0; t < 8; ++t) { a[t] = l0[t]; a[8 + t] = l1[t]; }
    return a;
}

// ---- CDNA5 async memory->LDS copy (16B per lane), tracked by ASYNCcnt ----------------------
__device__ __forceinline__ void async_copy16(_Float16* lds_dst, const _Float16* gsrc) {
    uint32_t lds_off = (uint32_t)(size_t)lds_dst;   // LDS aperture: addr[31:0] = LDS byte offset
    asm volatile("global_load_async_to_lds_b128 %0, %1, off"
                 :: "v"(lds_off), "v"(gsrc) : "memory");
}
__device__ __forceinline__ void wait_async0() {
    asm volatile("s_wait_asynccnt 0x0" ::: "memory");
}

// ---------------- Kernel 1: LayerNorm1 + roll(-4,-4) -> y (f32, rolled layout) --------------
__global__ __launch_bounds__(256) void k_ln_roll(const float* __restrict__ x,
                                                 const float* __restrict__ g,
                                                 const float* __restrict__ b,
                                                 float* __restrict__ y) {
    int token = blockIdx.x * 8 + (threadIdx.x >> 5);
    int lane = threadIdx.x & 31;
    int bimg = token >> 14, pix = token & 16383;
    int i = pix >> 7, j = pix & 127;
    int si = (i + 4) & 127, sj = (j + 4) & 127;
    const float* src = x + ((size_t)bimg * 16384 + si * 128 + sj) * 384;
    float vals[12], s = 0.f, ss = 0.f;
    #pragma unroll
    for (int t = 0; t < 12; ++t) {
        float v = src[lane + t * 32];
        vals[t] = v; s += v; ss += v * v;
    }
    s = wred_sum32(s); ss = wred_sum32(ss);
    float mu = s * (1.f / 384.f);
    float inv = rsqrtf(ss * (1.f / 384.f) - mu * mu + 1e-5f);
    float* dst = y + (size_t)token * 384;
    #pragma unroll
    for (int t = 0; t < 12; ++t) {
        int c = lane + t * 32;
        dst[c] = (vals[t] - mu) * inv * g[c] + b[c];
    }
}

// ---------------- Kernel 2: per-window dwconv3x3 + LN -> q,k (N,d) and v (d,N) f16 ----------
__device__ __forceinline__ void proj_one(const float* __restrict__ ybase, int wh, int ww,
                                         int ti, int tj, int t,
                                         const float* __restrict__ wgt,
                                         const float* __restrict__ g, const float* __restrict__ bb,
                                         float scale, _Float16* __restrict__ myrow,
                                         _Float16* __restrict__ outw, int vmode) {
    float s = 0.f, ss = 0.f;
    #pragma unroll 1
    for (int c = 0; c < 384; ++c) {
        float acc = 0.f;
        #pragma unroll
        for (int di = 0; di < 3; ++di) {
            int ii = ti + di - 1;
            if ((unsigned)ii < 8u) {
                #pragma unroll
                for (int dj = 0; dj < 3; ++dj) {
                    int jj = tj + dj - 1;
                    if ((unsigned)jj < 8u)
                        acc += wgt[(di * 3 + dj) * 384 + c] *
                               ybase[((size_t)(wh * 8 + ii) * 128 + (ww * 8 + jj)) * 384 + c];
                }
            }
        }
        myrow[c] = (_Float16)acc;
        s += acc; ss += acc * acc;
    }
    float mu = s * (1.f / 384.f);
    float inv = rsqrtf(ss * (1.f / 384.f) - mu * mu + 1e-5f);
    #pragma unroll 1
    for (int c = 0; c < 384; ++c) {
        float v = ((float)myrow[c] - mu) * inv * g[c] + bb[c];
        v *= scale;
        int h = c >> 5, d = c & 31;
        if (vmode == 0) outw[((size_t)h * 64 + t) * 32 + d] = (_Float16)v;   // (h,N,d)
        else            outw[((size_t)h * 32 + d) * 64 + t] = (_Float16)v;   // (h,d,N)
    }
}

__global__ __launch_bounds__(64) void k_qkv(const float* __restrict__ y,
                                            const float* __restrict__ wq, const float* __restrict__ gq, const float* __restrict__ bq,
                                            const float* __restrict__ wk, const float* __restrict__ gk, const float* __restrict__ bk,
                                            const float* __restrict__ wv, const float* __restrict__ gv, const float* __restrict__ bv,
                                            _Float16* __restrict__ q, _Float16* __restrict__ k, _Float16* __restrict__ vt) {
    __shared__ _Float16 sb[64 * 384];   // 48 KB, per-thread rows
    int win = blockIdx.x;
    int b = win >> 8, wh = (win >> 4) & 15, ww = win & 15;
    int t = threadIdx.x, ti = t >> 3, tj = t & 7;
    const float* ybase = y + (size_t)b * 16384 * 384;
    _Float16* myrow = sb + (size_t)t * 384;
    size_t wo = (size_t)win * 12 * 64 * 32;
    const float scale = 0.17677669529663687f; // 32^-0.5
    proj_one(ybase, wh, ww, ti, tj, t, wq, gq, bq, scale, myrow, q + wo, 0);
    proj_one(ybase, wh, ww, ti, tj, t, wk, gk, bk, 1.0f,  myrow, k + wo, 0);
    proj_one(ybase, wh, ww, ti, tj, t, wv, gv, bv, 1.0f,  myrow, vt + wo, 1);
}

// ---------------- Kernel 3: WMMA attention, one wave per (window, head) ---------------------
__global__ __launch_bounds__(32) void k_attn(const _Float16* __restrict__ q,
                                             const _Float16* __restrict__ k,
                                             const _Float16* __restrict__ vt,
                                             _Float16* __restrict__ E) {
    __shared__ _Float16 lp[64 * 72];   // P matrix round-trip, padded stride
    int bid = blockIdx.x;
    int win = bid / 12, head = bid % 12;
    int lane = threadIdx.x, lo = lane & 15, hi = lane >> 4;
    int wh = (win >> 4) & 15, ww = win & 15;

    const _Float16* qw = q  + ((size_t)win * 12 + head) * 64 * 32;
    const _Float16* kw = k  + ((size_t)win * 12 + head) * 64 * 32;
    const _Float16* vw = vt + ((size_t)win * 12 + head) * 32 * 64;

    // S = q * k^T : 4x4 tiles, K=32
    v16h aq[4];
    #pragma unroll
    for (int mt = 0; mt < 4; ++mt) aq[mt] = load_afrag(qw + (mt * 16 + lo) * 32, hi);
    v16h bk[4];
    #pragma unroll
    for (int nt = 0; nt < 4; ++nt) bk[nt] = *(const v16h*)(kw + (nt * 16 + lo) * 32 + 16 * hi);
    v8f c[4][4];
    #pragma unroll
    for (int mt = 0; mt < 4; ++mt)
        #pragma unroll
        for (int nt = 0; nt < 4; ++nt) { c[mt][nt] = {}; c[mt][nt] = wmma_f16(aq[mt], bk[nt], c[mt][nt]); }

    // shifted-window mask labels (computed analytically; H-WS=120, H-SHIFT=124)
    int clab[4];
    #pragma unroll
    for (int nt = 0; nt < 4; ++nt) {
        int tn = nt * 16 + lo;
        int gr = wh * 8 + (tn >> 3), gc = ww * 8 + (tn & 7);
        int rl = (gr < 120) ? 0 : ((gr < 124) ? 1 : 2);
        int cl = (gc < 120) ? 0 : ((gc < 124) ? 1 : 2);
        clab[nt] = rl * 3 + cl;
    }
    // masked softmax: row (mt,i,hi) spans 4 tiles x 16 lanes within the half-wave
    #pragma unroll
    for (int mt = 0; mt < 4; ++mt) {
        #pragma unroll
        for (int i = 0; i < 8; ++i) {
            int M = mt * 16 + i + 8 * hi;
            int gr = wh * 8 + (M >> 3), gc = ww * 8 + (M & 7);
            int rl = (gr < 120) ? 0 : ((gr < 124) ? 1 : 2);
            int cl = (gc < 120) ? 0 : ((gc < 124) ? 1 : 2);
            int rlab = rl * 3 + cl;
            float mx = -1e30f;
            #pragma unroll
            for (int nt = 0; nt < 4; ++nt) {
                if (rlab != clab[nt]) c[mt][nt][i] += -100.f;
                mx = fmaxf(mx, c[mt][nt][i]);
            }
            #pragma unroll
            for (int o = 8; o; o >>= 1) mx = fmaxf(mx, __shfl_xor(mx, o, 32));
            float s = 0.f;
            #pragma unroll
            for (int nt = 0; nt < 4; ++nt) {
                float e = __expf(c[mt][nt][i] - mx);
                c[mt][nt][i] = e; s += e;
            }
            #pragma unroll
            for (int o = 8; o; o >>= 1) s += __shfl_xor(s, o, 32);
            float inv = 1.f / s;
            #pragma unroll
            for (int nt = 0; nt < 4; ++nt) c[mt][nt][i] *= inv;
        }
    }
    // stage P in LDS, re-fragment as A-layout
    #pragma unroll
    for (int mt = 0; mt < 4; ++mt)
        #pragma unroll
        for (int nt = 0; nt < 4; ++nt)
            #pragma unroll
            for (int i = 0; i < 8; ++i) {
                int M = mt * 16 + i + 8 * hi, N = nt * 16 + lo;
                lp[M * 72 + N] = (_Float16)c[mt][nt][i];
            }
    __syncthreads();

    v8f o[4][2];
    #pragma unroll
    for (int mt = 0; mt < 4; ++mt)
        #pragma unroll
        for (int dt = 0; dt < 2; ++dt) o[mt][dt] = {};
    #pragma unroll
    for (int kt = 0; kt < 2; ++kt) {
        v16h ap[4];
        #pragma unroll
        for (int mt = 0; mt < 4; ++mt) {
            const _Float16* row = lp + (mt * 16 + lo) * 72 + kt * 32;
            v8h l0 = *(const v8h*)(row + 16 * hi);
            v8h l1 = *(const v8h*)(row + 16 * hi + 8);
            #pragma unroll
            for (int t = 0; t < 8; ++t) { ap[mt][t] = l0[t]; ap[mt][8 + t] = l1[t]; }
        }
        v16h bv[2];
        #pragma unroll
        for (int dt = 0; dt < 2; ++dt)
            bv[dt] = *(const v16h*)(vw + (dt * 16 + lo) * 64 + kt * 32 + 16 * hi);
        #pragma unroll
        for (int mt = 0; mt < 4; ++mt)
            #pragma unroll
            for (int dt = 0; dt < 2; ++dt) o[mt][dt] = wmma_f16(ap[mt], bv[dt], o[mt][dt]);
    }
    // write (win, N, C) with channel = head*32 + d
    #pragma unroll
    for (int mt = 0; mt < 4; ++mt)
        #pragma unroll
        for (int dt = 0; dt < 2; ++dt)
            #pragma unroll
            for (int i = 0; i < 8; ++i) {
                int M = mt * 16 + i + 8 * hi, dc = dt * 16 + lo;
                E[((size_t)win * 64 + M) * 384 + head * 32 + dc] = (_Float16)o[mt][dt][i];
            }
}

// ------------- Kernel 4: proj dwconv3x3 + GELU + window-reverse + roll(+4) + residual -------
__global__ __launch_bounds__(256) void k_projout(const _Float16* __restrict__ E,
                                                 const float* __restrict__ wproj,
                                                 const float* __restrict__ x,
                                                 float* __restrict__ x2) {
    int win = blockIdx.x;
    int b = win >> 8, wh = (win >> 4) & 15, ww = win & 15;
    int t = threadIdx.x & 63, cg = threadIdx.x >> 6;
    int ti = t >> 3, tj = t & 7;
    const _Float16* Ew = E + (size_t)win * 64 * 384;
    int gr = wh * 8 + ti, gc = ww * 8 + tj;
    int ro = (gr + 4) & 127, co = (gc + 4) & 127;
    size_t base = ((size_t)b * 16384 + ro * 128 + co) * 384;
    #pragma unroll 1
    for (int c = cg * 96; c < cg * 96 + 96; ++c) {
        float acc = 0.f;
        #pragma unroll
        for (int di = 0; di < 3; ++di) {
            int ii = ti + di - 1;
            if ((unsigned)ii < 8u) {
                #pragma unroll
                for (int dj = 0; dj < 3; ++dj) {
                    int jj = tj + dj - 1;
                    if ((unsigned)jj < 8u)
                        acc += wproj[(di * 3 + dj) * 384 + c] * (float)Ew[(ii * 8 + jj) * 384 + c];
                }
            }
        }
        x2[base + c] = x[base + c] + gelu_exact(acc);
    }
}

// ------------- Kernel 5: dwconv7x7 + bias + LN(1e-6) -> mln f16 -----------------------------
__global__ __launch_bounds__(256) void k_dwmlp(const float* __restrict__ x2,
                                               const float* __restrict__ dw, const float* __restrict__ db,
                                               const float* __restrict__ g, const float* __restrict__ bb,
                                               _Float16* __restrict__ mln) {
    int token = blockIdx.x * 8 + (threadIdx.x >> 5);
    int lane = threadIdx.x & 31;
    int bimg = token >> 14, pix = token & 16383;
    int i = pix >> 7, j = pix & 127;
    const float* xb = x2 + (size_t)bimg * 16384 * 384;
    float vals[12], s = 0.f, ss = 0.f;
    #pragma unroll 1
    for (int t = 0; t < 12; ++t) {
        int c = lane + t * 32;
        float acc = 0.f;
        #pragma unroll 1
        for (int di = 0; di < 7; ++di) {
            int ii = i + di - 3;
            if ((unsigned)ii < 128u) {
                #pragma unroll
                for (int dj = 0; dj < 7; ++dj) {
                    int jj = j + dj - 3;
                    if ((unsigned)jj < 128u)
                        acc += dw[(di * 7 + dj) * 384 + c] * xb[((size_t)ii * 128 + jj) * 384 + c];
                }
            }
        }
        acc += db[c];
        vals[t] = acc; s += acc; ss += acc * acc;
    }
    s = wred_sum32(s); ss = wred_sum32(ss);
    float mu = s * (1.f / 384.f);
    float inv = rsqrtf(ss * (1.f / 384.f) - mu * mu + 1e-6f);
    _Float16* dst = mln + (size_t)token * 384;
    #pragma unroll
    for (int t = 0; t < 12; ++t) {
        int c = lane + t * 32;
        dst[c] = (_Float16)((vals[t] - mu) * inv * g[c] + bb[c]);
    }
}

// ------------- weight transpose+cast: (R,Cc) f32 -> (Cc,R) f16 ------------------------------
__global__ void k_transpose_h(const float* __restrict__ src, _Float16* __restrict__ dst, int R, int Cc) {
    int i = blockIdx.x * blockDim.x + threadIdx.x;
    if (i < R * Cc) {
        int r = i / Cc, c = i % Cc;
        dst[(size_t)c * R + r] = (_Float16)src[i];
    }
}

// ------------- Double-buffered async WMMA GEMM main loop (shared by GEMM1/GEMM2) ------------
// A: (M x KD) f16 row-major; W: (N x KD) f16 n-major. Block tile 128x128, K-step 32.
// acc: per-wave 2x4 f32 fragments (wave covers rows wm*32.., cols wn*64..).
template <int KD>
__device__ __forceinline__ void gemm_mainloop(const _Float16* __restrict__ A,
                                              const _Float16* __restrict__ W,
                                              int bm, int bn, int tid,
                                              int wm, int wn, int lo, int hi,
                                              _Float16 (*sA)[128 * 32], _Float16 (*sB)[128 * 32],
                                              v8f acc[2][4]) {
    // prologue: async-fill buffer 0 with K-tile 0
    #pragma unroll
    for (int r = 0; r < 2; ++r) {
        int idx = tid + r * 256;
        int row = idx >> 2, col8 = (idx & 3) * 8;
        async_copy16(&sA[0][row * 32 + col8], A + (size_t)(bm * 128 + row) * KD + col8);
        async_copy16(&sB[0][row * 32 + col8], W + (size_t)(bn * 128 + row) * KD + col8);
    }
    const int KT = KD / 32;
    for (int kt = 0; kt < KT; ++kt) {
        int cur = kt & 1;
        wait_async0();        // my fills of buffer `cur` are complete
        __syncthreads();      // everyone's fills done; everyone done reading buffer 1-cur
        if (kt + 1 < KT) {
            int k0n = (kt + 1) * 32;
            #pragma unroll
            for (int r = 0; r < 2; ++r) {
                int idx = tid + r * 256;
                int row = idx >> 2, col8 = (idx & 3) * 8;
                async_copy16(&sA[1 - cur][row * 32 + col8], A + (size_t)(bm * 128 + row) * KD + k0n + col8);
                async_copy16(&sB[1 - cur][row * 32 + col8], W + (size_t)(bn * 128 + row) * KD + k0n + col8);
            }
            if (kt + 2 < KT) {  // speculative L2 prefetch two tiles ahead
                int k0p = (kt + 2) * 32;
                __builtin_prefetch(A + (size_t)(bm * 128 + (tid >> 1)) * KD + k0p, 0, 1);
                __builtin_prefetch(W + (size_t)(bn * 128 + (tid >> 1)) * KD + k0p, 0, 1);
            }
        }
        v16h af[2], bf[4];
        #pragma unroll
        for (int i = 0; i < 2; ++i) af[i] = load_afrag(&sA[cur][(wm * 32 + i * 16 + lo) * 32], hi);
        #pragma unroll
        for (int j = 0; j < 4; ++j) bf[j] = *(const v16h*)&sB[cur][(wn * 64 + j * 16 + lo) * 32 + 16 * hi];
        #pragma unroll
        for (int i = 0; i < 2; ++i)
            #pragma unroll
            for (int j = 0; j < 4; ++j) acc[i][j] = wmma_f16(af[i], bf[j], acc[i][j]);
        __syncthreads();      // done reading `cur` before it is refilled next+1 iteration
    }
}

// ------------- Kernel 6: GEMM1 (65536x384)@(384x1536) + bias + GELU -> hid f16 --------------
__global__ __launch_bounds__(256) void k_gemm1(const _Float16* __restrict__ A,
                                               const _Float16* __restrict__ W /* (1536,384) n-major */,
                                               const float* __restrict__ bias,
                                               _Float16* __restrict__ Hid) {
    __shared__ _Float16 sA[2][128 * 32];
    __shared__ _Float16 sB[2][128 * 32];
    int bm = blockIdx.x, bn = blockIdx.y;
    int tid = threadIdx.x;
    int wave = tid >> 5, lane = tid & 31, lo = lane & 15, hi = lane >> 4;
    int wm = wave >> 1, wn = wave & 1;
    v8f acc[2][4];
    #pragma unroll
    for (int i = 0; i < 2; ++i)
        #pragma unroll
        for (int j = 0; j < 4; ++j) acc[i][j] = {};
    gemm_mainloop<384>(A, W, bm, bn, tid, wm, wn, lo, hi, sA, sB, acc);
    #pragma unroll
    for (int i = 0; i < 2; ++i)
        #pragma unroll
        for (int j = 0; j < 4; ++j)
            #pragma unroll
            for (int r = 0; r < 8; ++r) {
                int gm = bm * 128 + wm * 32 + i * 16 + r + 8 * hi;
                int gn = bn * 128 + wn * 64 + j * 16 + lo;
                float v = gelu_exact(acc[i][j][r] + bias[gn]);
                Hid[(size_t)gm * 1536 + gn] = (_Float16)v;
            }
}

// ------------- Kernel 7: GEMM2 (65536x1536)@(1536x384) + bias, *gamma, + x2 -> out f32 ------
__global__ __launch_bounds__(256) void k_gemm2(const _Float16* __restrict__ A,
                                               const _Float16* __restrict__ W /* (384,1536) n-major */,
                                               const float* __restrict__ bias,
                                               const float* __restrict__ gamma,
                                               const float* __restrict__ x2,
                                               float* __restrict__ out) {
    __shared__ _Float16 sA[2][128 * 32];
    __shared__ _Float16 sB[2][128 * 32];
    int bm = blockIdx.x, bn = blockIdx.y;
    int tid = threadIdx.x;
    int wave = tid >> 5, lane = tid & 31, lo = lane & 15, hi = lane >> 4;
    int wm = wave >> 1, wn = wave & 1;
    v8f acc[2][4];
    #pragma unroll
    for (int i = 0; i < 2; ++i)
        #pragma unroll
        for (int j = 0; j < 4; ++j) acc[i][j] = {};
    gemm_mainloop<1536>(A, W, bm, bn, tid, wm, wn, lo, hi, sA, sB, acc);
    #pragma unroll
    for (int i = 0; i < 2; ++i)
        #pragma unroll
        for (int j = 0; j < 4; ++j)
            #pragma unroll
            for (int r = 0; r < 8; ++r) {
                int gm = bm * 128 + wm * 32 + i * 16 + r + 8 * hi;
                int gn = bn * 128 + wn * 64 + j * 16 + lo;
                size_t idx = (size_t)gm * 384 + gn;
                out[idx] = x2[idx] + gamma[gn] * (acc[i][j][r] + bias[gn]);
            }
}

extern "C" void kernel_launch(void* const* d_in, const int* in_sizes, int n_in,
                              void* d_out, int out_size, void* d_ws, size_t ws_size,
                              hipStream_t stream) {
    const float* x       = (const float*)d_in[0];
    const float* n1g     = (const float*)d_in[1];
    const float* n1b     = (const float*)d_in[2];
    const float* wq      = (const float*)d_in[3];
    const float* lnqg    = (const float*)d_in[4];
    const float* lnqb    = (const float*)d_in[5];
    const float* wk      = (const float*)d_in[6];
    const float* lnkg    = (const float*)d_in[7];
    const float* lnkb    = (const float*)d_in[8];
    const float* wv      = (const float*)d_in[9];
    const float* lnvg    = (const float*)d_in[10];
    const float* lnvb    = (const float*)d_in[11];
    const float* wproj   = (const float*)d_in[12];
    const float* dw_w    = (const float*)d_in[13];
    const float* dw_b    = (const float*)d_in[14];
    const float* mng     = (const float*)d_in[15];
    const float* mnb     = (const float*)d_in[16];
    const float* pw1_w   = (const float*)d_in[17];
    const float* pw1_b   = (const float*)d_in[18];
    const float* pw2_w   = (const float*)d_in[19];
    const float* pw2_b   = (const float*)d_in[20];
    const float* gamma   = (const float*)d_in[21];

    char* ws = (char*)d_ws;
    float*    y   = (float*)   (ws + 0);             // 100663296 B (reused by mln later)
    _Float16* q   = (_Float16*)(ws + 100663296);     // 50331648  (reused by hid later)
    _Float16* kk  = (_Float16*)(ws + 150994944);     // 50331648
    _Float16* vt  = (_Float16*)(ws + 201326592);     // 50331648
    _Float16* E   = (_Float16*)(ws + 251658240);     // 50331648
    float*    x2  = (float*)   (ws + 301989888);     // 100663296
    _Float16* w1t = (_Float16*)(ws + 402653184);     // 1179648
    _Float16* w2t = (_Float16*)(ws + 403832832);     // 1179648
    _Float16* mln = (_Float16*)(ws + 0);             // reuse y region
    _Float16* hid = (_Float16*)(ws + 100663296);     // reuse q/k/v/E region (exactly 201326592 B)

    k_transpose_h<<<(384 * 1536 + 255) / 256, 256, 0, stream>>>(pw1_w, w1t, 384, 1536);
    k_transpose_h<<<(1536 * 384 + 255) / 256, 256, 0, stream>>>(pw2_w, w2t, 1536, 384);
    k_ln_roll<<<8192, 256, 0, stream>>>(x, n1g, n1b, y);
    k_qkv<<<1024, 64, 0, stream>>>(y, wq, lnqg, lnqb, wk, lnkg, lnkb, wv, lnvg, lnvb, q, kk, vt);
    k_attn<<<12288, 32, 0, stream>>>(q, kk, vt, E);
    k_projout<<<1024, 256, 0, stream>>>(E, wproj, x, x2);
    k_dwmlp<<<8192, 256, 0, stream>>>(x2, dw_w, dw_b, mng, mnb, mln);
    k_gemm1<<<dim3(512, 12), 256, 0, stream>>>(mln, w1t, pw1_b, hid);
    k_gemm2<<<dim3(512, 3), 256, 0, stream>>>(hid, w2t, pw2_b, gamma, x2, (float*)d_out);
}
